// CrossAttention_85418309583262
// MI455X (gfx1250) — matrix operand order repaired and checked by reading, not compiled, not run on hardware
//
#include <hip/hip_runtime.h>
#include <hip/hip_bf16.h>

#define BATCH 128
#define SEQL  200
#define DIM   512
#define NHEAD 8
#define DKH   64
#define MROWS (BATCH * SEQL)      // 25600
#define QTILES 13                 // ceil(200/16)
#define LPAD  224                 // 200 padded to multiple of 32
#define NEGV  (-1e9f)

// attention LDS layout (dynamic)
#define KROW  72                  // Kbf row stride (bf16): 64 + 8 pad
#define VROW  232                 // Vt row stride (bf16): 224 + 8 pad
#define PROW  224                 // P row stride (bf16)
#define SM_K     (LPAD * KROW * 2)                    // 32256
#define SM_V     (DKH * VROW * 2)                     // 29696
#define SM_COLM  (LPAD * 4)                           // 896
#define SM_P     (8 * 16 * PROW * 2)                  // 57344
#define SM_TOTAL (SM_K + SM_V + SM_COLM + SM_P)       // 120192

typedef __attribute__((ext_vector_type(16))) __bf16 v16bf;
typedef __attribute__((ext_vector_type(8)))  __bf16 v8bf;
typedef __attribute__((ext_vector_type(4)))  __bf16 v4bf;
typedef __attribute__((ext_vector_type(8)))  float  v8f;

__device__ __forceinline__ v16bf pack16(v8bf lo, v8bf hi) {
    v16bf r;
#pragma unroll
    for (int t = 0; t < 8; ++t) { r[t] = lo[t]; r[t + 8] = hi[t]; }
    return r;
}

__device__ __forceinline__ v8f wmma_bf(v16bf a, v16bf b, v8f c) {
    return __builtin_amdgcn_wmma_f32_16x16x32_bf16(false, a, false, b,
                                                   (short)0, c, false, false);
}

// convert 8 consecutive f32 (16B-aligned) into fragment elements [base, base+8)
__device__ __forceinline__ void cvt8(const float* __restrict__ p, v16bf& f, int base) {
    const float4 a = *(const float4*)p;
    const float4 b = *(const float4*)(p + 4);
    f[base + 0] = (__bf16)a.x; f[base + 1] = (__bf16)a.y;
    f[base + 2] = (__bf16)a.z; f[base + 3] = (__bf16)a.w;
    f[base + 4] = (__bf16)b.x; f[base + 5] = (__bf16)b.y;
    f[base + 6] = (__bf16)b.z; f[base + 7] = (__bf16)b.w;
}

// ---------------------------------------------------------------------------
// prep: target = emb * (log!=0), source = emb * (log==0)
// ---------------------------------------------------------------------------
__global__ __launch_bounds__(256) void prep_kernel(
    const float* __restrict__ emb, const int* __restrict__ logs,
    float* __restrict__ target, float* __restrict__ source)
{
    size_t i = (size_t)blockIdx.x * 256 + threadIdx.x;   // over B*L*D
    size_t row = i >> 9;                                 // /512
    float e = emb[i];
    bool tl = (logs[row] == 0);
    target[i] = tl ? 0.0f : e;
    source[i] = tl ? e : 0.0f;
}

// ---------------------------------------------------------------------------
// layernorm: out = LN(addbase ? base+x : x) * g + b   [* keep]
// ---------------------------------------------------------------------------
__global__ __launch_bounds__(256) void layernorm_kernel(
    const float* __restrict__ x, const float* __restrict__ base,
    const float* __restrict__ g, const float* __restrict__ bta,
    const int* __restrict__ keep, float* __restrict__ out, int addbase)
{
    __shared__ float redS[8], redQ[8];
    int row = blockIdx.x;
    int tid = threadIdx.x;
    int lane = tid & 31, wave = tid >> 5;
    size_t roff = (size_t)row * DIM;
    int c0 = tid, c1 = tid + 256;

    float v0 = x[roff + c0], v1 = x[roff + c1];
    if (addbase) { v0 += base[roff + c0]; v1 += base[roff + c1]; }
    float s = v0 + v1, q = v0 * v0 + v1 * v1;
#pragma unroll
    for (int off = 16; off; off >>= 1) {
        s += __shfl_xor(s, off, 32);
        q += __shfl_xor(q, off, 32);
    }
    if (lane == 0) { redS[wave] = s; redQ[wave] = q; }
    __syncthreads();
    float S = 0.f, Q = 0.f;
#pragma unroll
    for (int w = 0; w < 8; ++w) { S += redS[w]; Q += redQ[w]; }
    float mean = S * (1.0f / DIM);
    float var  = Q * (1.0f / DIM) - mean * mean;
    float rstd = rsqrtf(var + 1e-8f);
    float kf = keep ? ((keep[row] != 0) ? 1.0f : 0.0f) : 1.0f;
    out[roff + c0] = ((v0 - mean) * rstd * g[c0] + bta[c0]) * kf;
    out[roff + c1] = ((v1 - mean) * rstd * g[c1] + bta[c1]) * kf;
}

// ---------------------------------------------------------------------------
// GEMM: C[M,512] = epilogue(A[M,512] @ W[512,512] + bias)
// block tile 128x128, 8 waves each 32x64 (2x4 wmma), K step 32, bf16 staging
// ---------------------------------------------------------------------------
#define GTM 128
#define GTN 128
#define GTK 32
#define LDP 40   // padded LDS row stride (bf16), keeps 16B alignment

template <int RELU, int RESID, int KEEP>
__global__ __launch_bounds__(256) void gemm_bf16_kernel(
    const float* __restrict__ A, const float* __restrict__ W,
    const float* __restrict__ bias, float* __restrict__ C,
    const float* __restrict__ resid, const int* __restrict__ keep)
{
    __shared__ __bf16 sA[GTM][LDP];
    __shared__ __bf16 sW[GTN][LDP];   // W tile transposed: [n][k]

    int tid = threadIdx.x;
    int lane = tid & 31, wave = tid >> 5;
    int bm = blockIdx.x * GTM;
    int bn = blockIdx.y * GTN;
    int wm = (wave & 3) * 32;
    int wn = (wave >> 2) * 64;
    int hsel  = (lane >= 16) ? 8 : 0;
    int khalf = (lane >= 16) ? 16 : 0;

    v8f acc[2][4] = {};

    for (int kk = 0; kk < DIM; kk += GTK) {
#pragma unroll
        for (int t = tid; t < (GTM * GTK) / 4; t += 256) {   // A tile
            int c4 = (t & 7) * 4;
            int r  = t >> 3;
            const float4 a4 = *(const float4*)&A[(size_t)(bm + r) * DIM + kk + c4];
            v4bf cv;
            cv[0] = (__bf16)a4.x; cv[1] = (__bf16)a4.y;
            cv[2] = (__bf16)a4.z; cv[3] = (__bf16)a4.w;
            *(v4bf*)&sA[r][c4] = cv;
        }
#pragma unroll
        for (int t = tid; t < (GTK * GTN) / 4; t += 256) {   // W tile (transposed)
            int n4 = (t & 31) * 4;
            int kr = t >> 5;
            const float4 w4 = *(const float4*)&W[(size_t)(kk + kr) * DIM + bn + n4];
            sW[n4 + 0][kr] = (__bf16)w4.x;
            sW[n4 + 1][kr] = (__bf16)w4.y;
            sW[n4 + 2][kr] = (__bf16)w4.z;
            sW[n4 + 3][kr] = (__bf16)w4.w;
        }
        __syncthreads();

        v16bf af[2], bf[4];
#pragma unroll
        for (int i = 0; i < 2; ++i) {
            int r = wm + i * 16 + (lane & 15);
            af[i] = pack16(*(const v8bf*)&sA[r][hsel],
                           *(const v8bf*)&sA[r][hsel + 16]);
        }
#pragma unroll
        for (int j = 0; j < 4; ++j) {
            int n = wn + j * 16 + (lane & 15);
            bf[j] = pack16(*(const v8bf*)&sW[n][hsel],
                           *(const v8bf*)&sW[n][hsel + 16]);
        }
#pragma unroll
        for (int i = 0; i < 2; ++i)
#pragma unroll
            for (int j = 0; j < 4; ++j)
                acc[i][j] = wmma_bf(af[i], bf[j], acc[i][j]);
        __syncthreads();
    }

#pragma unroll
    for (int i = 0; i < 2; ++i) {
#pragma unroll
        for (int j = 0; j < 4; ++j) {
            int ncol = bn + wn + j * 16 + (lane & 15);
            float bv = bias[ncol];
#pragma unroll
            for (int r = 0; r < 8; ++r) {
                int mrow = bm + wm + i * 16 + r + ((lane >> 4) << 3);
                float val = acc[i][j][r] + bv;
                if (RELU)  val = fmaxf(val, 0.0f);
                if (RESID) val += resid[(size_t)mrow * DIM + ncol];
                if (KEEP)  val *= (keep[mrow] != 0) ? 1.0f : 0.0f;
                C[(size_t)mrow * DIM + ncol] = val;
            }
        }
    }
}

// ---------------------------------------------------------------------------
// attention v2: one 256-thread block per (batch, head)
// K and V^T staged once in dynamic LDS (bf16); 8 waves sweep 13 query tiles;
// scores + softmax fully register-resident; P transposed through per-wave LDS.
// ---------------------------------------------------------------------------
__global__ __launch_bounds__(256) void attn_kernel(
    const float* __restrict__ q, const float* __restrict__ k,
    const float* __restrict__ v, const int* __restrict__ logs,
    const int* __restrict__ alls, float* __restrict__ xout)
{
    extern __shared__ char smem_raw[];
    __bf16* Kbf  = (__bf16*)smem_raw;                       // [LPAD][KROW]
    __bf16* Vt   = (__bf16*)(smem_raw + SM_K);              // [DKH][VROW]
    float*  colm = (float*)(smem_raw + SM_K + SM_V);        // [LPAD]
    __bf16* Pall = (__bf16*)(smem_raw + SM_K + SM_V + SM_COLM);

    int h = blockIdx.x % NHEAD;
    int b = blockIdx.x / NHEAD;
    int tid = threadIdx.x;
    int lane = tid & 31, wave = tid >> 5;
    int hsel  = (lane >= 16) ? 8 : 0;
    int khalf = (lane >= 16) ? 16 : 0;
    __bf16* Pw = Pall + wave * 16 * PROW;

    // ---- stage K [key][d] and V^T [d][key] as bf16, plus column mask ----
#pragma unroll
    for (int t = tid; t < LPAD * 16; t += 256) {            // 14 iters, float4
        int key = t >> 4;
        int d4  = (t & 15) * 4;
        int kc  = key > (SEQL - 1) ? (SEQL - 1) : key;
        const float4 kv = *(const float4*)&k[((size_t)b * SEQL + kc) * DIM
                                             + h * DKH + d4];
        v4bf cv;
        cv[0] = (__bf16)kv.x; cv[1] = (__bf16)kv.y;
        cv[2] = (__bf16)kv.z; cv[3] = (__bf16)kv.w;
        *(v4bf*)&Kbf[key * KROW + d4] = cv;
    }
#pragma unroll
    for (int t = tid; t < LPAD * 16; t += 256) {
        int key = t >> 4;
        int d4  = (t & 15) * 4;
        int kc  = key > (SEQL - 1) ? (SEQL - 1) : key;      // P there is 0
        const float4 vv = *(const float4*)&v[((size_t)b * SEQL + kc) * DIM
                                             + h * DKH + d4];
        Vt[(d4 + 0) * VROW + key] = (__bf16)vv.x;
        Vt[(d4 + 1) * VROW + key] = (__bf16)vv.y;
        Vt[(d4 + 2) * VROW + key] = (__bf16)vv.z;
        Vt[(d4 + 3) * VROW + key] = (__bf16)vv.w;
    }
    if (tid < LPAD) {
        int ok = (tid < SEQL) && (logs[b * SEQL + tid] == 0)
                              && (alls[b * SEQL + tid] != 0);
        colm[tid] = ok ? 1.0f : 0.0f;
    }
    __syncthreads();

    // ---- per-wave query tiles (uniform loop -> EXEC all-ones at WMMAs) ----
    for (int qt = wave; qt < QTILES; qt += 8) {
        // A-fragments from q (16 query rows, dk=64 -> 2 K-steps)
        int qrow = qt * 16 + (lane & 15);
        int qrc = qrow > (SEQL - 1) ? (SEQL - 1) : qrow;
        size_t qbase = ((size_t)b * SEQL + qrc) * DIM + h * DKH;
        v16bf aq[2];
#pragma unroll
        for (int s2 = 0; s2 < 2; ++s2) {
            cvt8(&q[qbase + hsel + s2 * 32],      aq[s2], 0);
            cvt8(&q[qbase + hsel + 16 + s2 * 32], aq[s2], 8);
        }

        // scores: 14 key tiles, register-resident
        v8f accS[14];
#pragma unroll
        for (int j = 0; j < 14; ++j) {
            int kg = j * 16 + (lane & 15);
            v8f a = {};
#pragma unroll
            for (int s2 = 0; s2 < 2; ++s2) {
                v16bf bk = pack16(
                    *(const v8bf*)&Kbf[kg * KROW + s2 * 32 + hsel],
                    *(const v8bf*)&Kbf[kg * KROW + s2 * 32 + hsel + 16]);
                a = wmma_bf(aq[s2], bk, a);
            }
            accS[j] = a;
        }

        // mask + scale (branchless; colm hoisted per tile)
        float mx[8];
#pragma unroll
        for (int r = 0; r < 8; ++r) mx[r] = -3.0e38f;
#pragma unroll
        for (int j = 0; j < 14; ++j) {
            int kg = j * 16 + (lane & 15);
            float cm = colm[kg];
#pragma unroll
            for (int r = 0; r < 8; ++r) {
                int qg = qt * 16 + r + ((lane >> 4) << 3);
                float sc = accS[j][r] * 0.125f;
                float val = ((kg <= qg) & (cm > 0.5f)) ? sc : NEGV;
                accS[j][r] = val;
                mx[r] = fmaxf(mx[r], val);
            }
        }
        // row max across the 16-lane half
#pragma unroll
        for (int r = 0; r < 8; ++r)
#pragma unroll
            for (int o = 1; o < 16; o <<= 1)
                mx[r] = fmaxf(mx[r], __shfl_xor(mx[r], o, 32));
        // row sums (masked entries excluded: exp(NEG-mx) == 0 in fp32 anyway)
        float sm[8];
#pragma unroll
        for (int r = 0; r < 8; ++r) sm[r] = 0.0f;
#pragma unroll
        for (int j = 0; j < 14; ++j)
#pragma unroll
            for (int r = 0; r < 8; ++r) {
                float val = accS[j][r];
                sm[r] += (val > -5.0e8f) ? __expf(val - mx[r]) : 0.0f;
            }
        float inv[8];
#pragma unroll
        for (int r = 0; r < 8; ++r) {
#pragma unroll
            for (int o = 1; o < 16; o <<= 1)
                sm[r] += __shfl_xor(sm[r], o, 32);
            inv[r] = 1.0f / sm[r];   // all-masked row: P forced to 0 below
        }
        // P = softmax * mask  -> per-wave LDS (bf16), transposing layout
#pragma unroll
        for (int j = 0; j < 14; ++j) {
            int kg = j * 16 + (lane & 15);
#pragma unroll
            for (int r = 0; r < 8; ++r) {
                int m = r + ((lane >> 4) << 3);
                float val = accS[j][r];
                float p = (val > -5.0e8f) ? __expf(val - mx[r]) * inv[r] : 0.0f;
                Pw[m * PROW + kg] = (__bf16)p;
            }
        }
        // same-wave LDS ops are in-order: no barrier needed before reading P

        // O = P @ V : 7 K-steps x 4 output col tiles
        v8f accO[4] = {};
        int prow = lane & 15;
        for (int kk = 0; kk < LPAD; kk += 32) {
            v16bf pa = pack16(*(const v8bf*)&Pw[prow * PROW + kk + hsel],
                              *(const v8bf*)&Pw[prow * PROW + kk + hsel + 16]);
#pragma unroll
            for (int n = 0; n < 4; ++n) {
                int d = n * 16 + (lane & 15);
                v16bf vb = pack16(*(const v8bf*)&Vt[d * VROW + kk + hsel],
                                  *(const v8bf*)&Vt[d * VROW + kk + hsel + 16]);
                accO[n] = wmma_bf(pa, vb, accO[n]);
            }
        }
#pragma unroll
        for (int n = 0; n < 4; ++n)
#pragma unroll
            for (int r = 0; r < 8; ++r) {
                int m = r + ((lane >> 4) << 3);
                int qg = qt * 16 + m;
                if (qg < SEQL)
                    xout[((size_t)b * SEQL + qg) * DIM + h * DKH
                         + n * 16 + (lane & 15)] = accO[n][r];
            }
    }
}

// ---------------------------------------------------------------------------
extern "C" void kernel_launch(void* const* d_in, const int* in_sizes, int n_in,
                              void* d_out, int out_size, void* d_ws, size_t ws_size,
                              hipStream_t stream) {
    (void)in_sizes; (void)n_in; (void)out_size; (void)ws_size;
    const int*   logs = (const int*)d_in[0];
    const float* emb  = (const float*)d_in[1];
    const int*   alls = (const int*)d_in[2];
    const float* Wq = (const float*)d_in[3];
    const float* bq = (const float*)d_in[4];
    const float* Wk = (const float*)d_in[5];
    const float* bk = (const float*)d_in[6];
    const float* Wv = (const float*)d_in[7];
    const float* bv = (const float*)d_in[8];
    const float* lag = (const float*)d_in[9];
    const float* lab = (const float*)d_in[10];
    const float* lfg = (const float*)d_in[11];
    const float* lfb = (const float*)d_in[12];
    const float* W1 = (const float*)d_in[13];
    const float* b1 = (const float*)d_in[14];
    const float* W2 = (const float*)d_in[15];
    const float* b2 = (const float*)d_in[16];
    const float* llg = (const float*)d_in[17];
    const float* llb = (const float*)d_in[18];

    const size_t slab = (size_t)MROWS * DIM;
    float* ws     = (float*)d_ws;
    float* target = ws + 0 * slab;
    float* source = ws + 1 * slab;
    float* qn     = ws + 2 * slab;   // reused as FFN hidden
    float* qb     = ws + 3 * slab;
    float* kb     = ws + 4 * slab;
    float* vb     = ws + 5 * slab;
    float* xb     = ws + 6 * slab;
    float* hb     = qn;              // alias: qn dead after Q projection

    // raise dynamic-LDS cap for the attention kernel (capture-safe, idempotent)
    (void)hipFuncSetAttribute((const void*)attn_kernel,
                              hipFuncAttributeMaxDynamicSharedMemorySize,
                              SM_TOTAL);

    dim3 ggrid(MROWS / GTM, DIM / GTN);   // (200, 4)

    prep_kernel<<<(unsigned)(slab / 256), 256, 0, stream>>>(emb, logs, target, source);

    for (int i = 0; i < 2; ++i) {
        size_t wo = (size_t)i * DIM * DIM;
        size_t vo = (size_t)i * DIM;
        layernorm_kernel<<<MROWS, 256, 0, stream>>>(target, nullptr, lag + vo,
                                                    lab + vo, nullptr, qn, 0);
        gemm_bf16_kernel<0,0,0><<<ggrid, 256, 0, stream>>>(qn, Wq + wo, bq + vo, qb,
                                                           nullptr, nullptr);
        gemm_bf16_kernel<0,0,0><<<ggrid, 256, 0, stream>>>(source, Wk + wo, bk + vo,
                                                           kb, nullptr, nullptr);
        gemm_bf16_kernel<0,0,0><<<ggrid, 256, 0, stream>>>(source, Wv + wo, bv + vo,
                                                           vb, nullptr, nullptr);
        attn_kernel<<<BATCH * NHEAD, 256, SM_TOTAL, stream>>>(qb, kb, vb, logs,
                                                              alls, xb);
        layernorm_kernel<<<MROWS, 256, 0, stream>>>(xb, target, lfg + vo, lfb + vo,
                                                    nullptr, target, i == 0 ? 0 : 1);
        gemm_bf16_kernel<1,0,0><<<ggrid, 256, 0, stream>>>(target, W1 + wo, b1 + vo,
                                                           hb, nullptr, nullptr);
        gemm_bf16_kernel<0,1,1><<<ggrid, 256, 0, stream>>>(hb, W2 + wo, b2 + vo,
                                                           target, target, logs);
    }
    layernorm_kernel<<<MROWS, 256, 0, stream>>>(target, nullptr, llg, llb, logs,
                                                (float*)d_out, 0);
}
